// MLADecodeBenchmarker_46737834115449
// MI455X (gfx1250) — matrix operand order repaired and checked by reading, not compiled, not run on hardware
//
#include <hip/hip_runtime.h>
#include <hip/hip_bf16.h>

// CDNA5 / gfx1250: wave32, WMMA bf16 16x16x32 with f32 accumulate.
typedef __attribute__((ext_vector_type(16))) __bf16 v16bf;
typedef __attribute__((ext_vector_type(8)))  __bf16 v8bf;
typedef __attribute__((ext_vector_type(8)))  float  v8f;

#define WMMA_BF16(acc, a, b)                                                   \
    (acc) = __builtin_amdgcn_wmma_f32_16x16x32_bf16(false, (a), false, (b),    \
                                                    (short)0, (acc), false, false)

// ---------------------------------------------------------------------------
// Elementwise f32 -> bf16 stream conversion. n must be a multiple of 8.
// ---------------------------------------------------------------------------
__global__ __launch_bounds__(256)
void cvt_f32_bf16(const float* __restrict__ in, __bf16* __restrict__ out, long n) {
    long i = ((long)blockIdx.x * 256 + threadIdx.x) * 8;
    if (i >= n) return;
    const float4 f0 = *(const float4*)(in + i);
    const float4 f1 = *(const float4*)(in + i + 4);
    v8bf o;
    o[0] = (__bf16)f0.x; o[1] = (__bf16)f0.y; o[2] = (__bf16)f0.z; o[3] = (__bf16)f0.w;
    o[4] = (__bf16)f1.x; o[5] = (__bf16)f1.y; o[6] = (__bf16)f1.z; o[7] = (__bf16)f1.w;
    *(v8bf*)(out + i) = o;
}

// ---------------------------------------------------------------------------
// Fragment helpers (wave32 WMMA operand layouts).
// ---------------------------------------------------------------------------
static __device__ __forceinline__ v16bf ld_a16(const __bf16* p) {
    // A 16x32 bf16: lane K-chunk is 16 contiguous elems (k = 16*half + s)
    const v8bf x0 = *(const v8bf*)(p);
    const v8bf x1 = *(const v8bf*)(p + 8);
    return __builtin_shufflevector(x0, x1, 0, 1, 2, 3, 4, 5, 6, 7,
                                           8, 9, 10, 11, 12, 13, 14, 15);
}
static __device__ __forceinline__ v16bf ld_b16(const __bf16* p) {
    const v8bf x0 = *(const v8bf*)(p);
    const v8bf x1 = *(const v8bf*)(p + 8);
    return __builtin_shufflevector(x0, x1, 0, 1, 2, 3, 4, 5, 6, 7,
                                           8, 9, 10, 11, 12, 13, 14, 15);
}

struct Frags {                 // one K=32 pipeline stage: 2 N-strips + 4 M-tiles
    v16bf b0, b1;
    v16bf a0, a1, a2, a3;
};

template <bool BT>
static __device__ __forceinline__
Frags load_frags(const __bf16* __restrict__ A, const __bf16* __restrict__ B,
                 long lda, long ldb, int n0, int r, int half, int k0) {
    Frags f;
    if (BT) {
        const __bf16* bp = B + (long)(n0 + r) * ldb + k0 + half * 16;
        f.b0 = ld_b16(bp);
        f.b1 = ld_b16(bp + 16 * ldb);
    } else {
        const __bf16* bp = B + (long)(k0 + half * 16) * ldb + (n0 + r);
        #pragma unroll
        for (int j = 0; j < 16; ++j) {         // lane-strided, contiguous across lanes
            f.b0[j] = bp[(long)j * ldb];
            f.b1[j] = bp[(long)j * ldb + 16];
        }
    }
    const __bf16* ap = A + (long)r * lda + k0 + half * 16;
    f.a0 = ld_a16(ap);
    f.a1 = ld_a16(ap + 16 * lda);
    f.a2 = ld_a16(ap + 32 * lda);
    f.a3 = ld_a16(ap + 48 * lda);
    return f;
}

static __device__ __forceinline__ void do_wmma(v8f acc[8], const Frags& f) {
    WMMA_BF16(acc[0], f.a0, f.b0);
    WMMA_BF16(acc[1], f.a1, f.b0);
    WMMA_BF16(acc[2], f.a2, f.b0);
    WMMA_BF16(acc[3], f.a3, f.b0);
    WMMA_BF16(acc[4], f.a0, f.b1);
    WMMA_BF16(acc[5], f.a1, f.b1);
    WMMA_BF16(acc[6], f.a2, f.b1);
    WMMA_BF16(acc[7], f.a3, f.b1);
}

// ---------------------------------------------------------------------------
// Strided-batched bf16 GEMM, M = 64 always.
//   BT == true :  C[m,n] = scale * sum_k A[m,k] * B[n,k]
//   BT == false:  C[m,n] = scale * sum_k A[m,k] * B[k,n]
// OBF: store C as bf16 (else f32).
// Each wave computes a 64(M) x 32(N) tile: 8 accumulators, software-pipelined
// K-loop (double-buffered fragments, loads issued a stage ahead of consumers).
// Block = 128 threads = 4 waves = 128 N columns. grid.y = batch.
// Requires: K % 64 == 0, K >= 128, N % 32 == 0.
// ---------------------------------------------------------------------------
template <bool BT, bool OBF>
__global__ __launch_bounds__(128)
void gemm_wmma_bf16(const __bf16* __restrict__ A, const __bf16* __restrict__ B,
                    void* __restrict__ Cv,
                    int N, int K,
                    long lda, long ldb, long ldc,
                    long sA, long sB, long sC,
                    float scale) {
    const int lane = threadIdx.x & 31;
    const int wave = threadIdx.x >> 5;
    const int n0   = (blockIdx.x * 4 + wave) * 32;
    if (n0 >= N) return;                       // wave-uniform: EXEC stays all-ones

    const long batch = blockIdx.y;
    A += batch * sA;
    B += batch * sB;

    const int r    = lane & 15;
    const int half = lane >> 4;

    v8f acc[8] = {};

    // stage 0 preload, then: load(k+32) | wmma(k) | load(k+64) | wmma(k+32)
    Frags cur = load_frags<BT>(A, B, lda, ldb, n0, r, half, 0);
    int k0 = 0;
    for (; k0 < K - 64; k0 += 64) {
        Frags nxt = load_frags<BT>(A, B, lda, ldb, n0, r, half, k0 + 32);
        do_wmma(acc, cur);
        cur = load_frags<BT>(A, B, lda, ldb, n0, r, half, k0 + 64);
        do_wmma(acc, nxt);
    }
    {
        Frags nxt = load_frags<BT>(A, B, lda, ldb, n0, r, half, K - 32);
        do_wmma(acc, cur);
        do_wmma(acc, nxt);
    }

    // ---- Store: D VGPR rr -> row (rr + half*8), col n0 + s*16 + r
    #pragma unroll
    for (int s = 0; s < 2; ++s) {
        #pragma unroll
        for (int m = 0; m < 4; ++m) {
            #pragma unroll
            for (int rr = 0; rr < 8; ++rr) {
                const long idx = batch * sC + (long)(m * 16 + half * 8 + rr) * ldc
                               + n0 + s * 16 + r;
                const float v = acc[s * 4 + m][rr] * scale;
                if (OBF) ((__bf16*)Cv)[idx] = (__bf16)v;
                else     ((float*)Cv)[idx]  = v;
            }
        }
    }
}

// ---------------------------------------------------------------------------
// Row-wise softmax: f32 scores in, bf16 probs out. One 256-thread block / row.
// ---------------------------------------------------------------------------
__global__ __launch_bounds__(256)
void softmax_rows(const float* __restrict__ scores, __bf16* __restrict__ probs,
                  int cols) {
    __shared__ float red[256];
    const float* row  = scores + (long)blockIdx.x * cols;
    __bf16*      orow = probs  + (long)blockIdx.x * cols;
    const int t = threadIdx.x;

    float m = -3.402823466e38f;
    for (int i = t; i < cols; i += 256) m = fmaxf(m, row[i]);
    red[t] = m; __syncthreads();
    for (int s = 128; s > 0; s >>= 1) {
        if (t < s) red[t] = fmaxf(red[t], red[t + s]);
        __syncthreads();
    }
    m = red[0]; __syncthreads();

    float e[4];                                 // cols == 1024, 4 per thread
    float sum = 0.0f;
    #pragma unroll
    for (int j = 0; j < 4; ++j) {
        e[j] = __expf(row[t + j * 256] - m);
        sum += e[j];
    }
    red[t] = sum; __syncthreads();
    for (int s = 128; s > 0; s >>= 1) {
        if (t < s) red[t] += red[t + s];
        __syncthreads();
    }
    const float inv = 1.0f / red[0];
    #pragma unroll
    for (int j = 0; j < 4; ++j) orow[t + j * 256] = (__bf16)(e[j] * inv);
}

// ---------------------------------------------------------------------------
// Host-side orchestration (absorbed-MLA decode pipeline).
// ---------------------------------------------------------------------------
extern "C" void kernel_launch(void* const* d_in, const int* in_sizes, int n_in,
                              void* d_out, int out_size, void* d_ws, size_t ws_size,
                              hipStream_t stream) {
    (void)in_sizes; (void)n_in; (void)out_size; (void)ws_size;

    // B=64, D_MODEL=4096, D_C=512, H=64, S=1024, HEAD_DIM=64
    const float* x      = (const float*)d_in[0];   // [64, 4096]
    const float* w_dq   = (const float*)d_in[1];   // [512, 4096]
    const float* w_qk   = (const float*)d_in[2];   // [32768, 512]
    const float* w_uv   = (const float*)d_in[3];   // [4096, 512]
    const float* seq_kv = (const float*)d_in[4];   // [64, 1024, 512]
    float* out = (float*)d_out;                    // [64, 4096]

    char* w = (char*)d_ws;
    const long MB = 1l << 20;
    __bf16* kv_bf   = (__bf16*)(w);                //  64 MB [0,64)
    __bf16* wqk_bf  = (__bf16*)(w +  64 * MB);     //  32 MB [64,96)  — dead after GEMM-2
    float*  scores  = (float*) (w +  64 * MB);     //  16 MB [64,80)  (aliases wqk_bf)
    __bf16* attn_bf = (__bf16*)(w +  80 * MB);     //   8 MB [80,88)  (aliases wqk_bf)
    __bf16* wdq_bf  = (__bf16*)(w +  96 * MB);     //   4 MB
    __bf16* wuv_bf  = (__bf16*)(w + 100 * MB);     //   4 MB
    __bf16* x_bf    = (__bf16*)(w + 104 * MB);     // 0.5 MB
    __bf16* qc_bf   = (__bf16*)(w + 105 * MB);     //  64 KB
    __bf16* qlat_bf = (__bf16*)(w + 106 * MB);     //   4 MB
    __bf16* olat_bf = (__bf16*)(w + 110 * MB);     //   4 MB  (total 114 MB)

    const dim3 blk(128);
    const dim3 cblk(256);
    auto cgrid = [](long n) { return dim3((unsigned)(n / 8 / 256)); };

    // 0) stream-convert operands to bf16 (pure bandwidth)
    cvt_f32_bf16<<<cgrid(33554432), cblk, 0, stream>>>(seq_kv, kv_bf, 33554432);
    cvt_f32_bf16<<<cgrid(16777216), cblk, 0, stream>>>(w_qk, wqk_bf, 16777216);
    cvt_f32_bf16<<<cgrid(2097152),  cblk, 0, stream>>>(w_dq, wdq_bf, 2097152);
    cvt_f32_bf16<<<cgrid(2097152),  cblk, 0, stream>>>(w_uv, wuv_bf, 2097152);
    cvt_f32_bf16<<<cgrid(262144),   cblk, 0, stream>>>(x, x_bf, 262144);

    // 1) q_c = x @ w_dq^T                 [64,512]  K=4096
    gemm_wmma_bf16<true, true><<<dim3(512 / 128, 1), blk, 0, stream>>>(
        x_bf, wdq_bf, qc_bf, 512, 4096, 4096, 4096, 512, 0, 0, 0, 1.0f);

    // 2) q_lat = q_c @ w_qk^T             [64,32768] K=512
    gemm_wmma_bf16<true, true><<<dim3(32768 / 128, 1), blk, 0, stream>>>(
        qc_bf, wqk_bf, qlat_bf, 32768, 512, 512, 512, 32768, 0, 0, 0, 1.0f);

    // 3) scores[b] = q_lat[b] @ seq_kv[b]^T * 64   [H=64, S=1024] per b, K=512
    //    (reference divides by SCALE = D_MODEL^-0.5, i.e. multiplies by 64)
    gemm_wmma_bf16<true, false><<<dim3(1024 / 128, 64), blk, 0, stream>>>(
        qlat_bf, kv_bf, scores, 1024, 512, 512, 512, 1024,
        32768, 524288, 65536, 64.0f);

    // 4) softmax over S for each (b,h) row; emit bf16 probs
    softmax_rows<<<dim3(64 * 64), dim3(256), 0, stream>>>(scores, attn_bf, 1024);

    // 5) o_lat[b] = attn[b] @ seq_kv[b]   [H=64, 512] per b, K=1024 (A@B form)
    gemm_wmma_bf16<false, true><<<dim3(512 / 128, 64), blk, 0, stream>>>(
        attn_bf, kv_bf, olat_bf, 512, 1024, 1024, 512, 512,
        65536, 524288, 32768, 1.0f);

    // 6) out[:, h*64:(h+1)*64] = o_lat[:,h,:] @ w_uv_h^T   [64,64] per h, K=512
    //    N=64 < 128: waves 2,3 exit via the wave-uniform guard.
    gemm_wmma_bf16<true, false><<<dim3(1, 64), blk, 0, stream>>>(
        olat_bf, wuv_bf, out, 64, 512, 32768, 512, 4096,
        512, 32768, 64, 1.0f);
}
